// Attention2D_47132971107192
// MI455X (gfx1250) — matrix operand order repaired and checked by reading, not compiled, not run on hardware
//
#include <hip/hip_runtime.h>
#include <hip/hip_bf16.h>

#define NTOK 4096           // H*W
#define CCH  256            // C
#define CDM  32             // C/8
#define BB   8
#define MTOT (BB * NTOK)    // 32768 total tokens

typedef _Float16 h16;
typedef __attribute__((ext_vector_type(16))) _Float16 v16h;
typedef __attribute__((ext_vector_type(8)))  _Float16 v8h;
typedef __attribute__((ext_vector_type(4)))  _Float16 v4h;
typedef __attribute__((ext_vector_type(8)))  float    v8f;
typedef __attribute__((ext_vector_type(4)))  float    v4f;

static __device__ __forceinline__ v8f wmma_f16(v16h a, v16h b, v8f c) {
  // (neg_a, A, neg_b, B, c_mod, C, reuse_a, reuse_b)
  return __builtin_amdgcn_wmma_f32_16x16x32_f16(false, a, false, b, (short)0, c,
                                                false, false);
}

// ---- Fragment loaders (layouts per CDNA5 ISA, wave32) ----------
// A 16x32 f16: lane L -> row m=L&15; ko=8*(L>>4); elems 0..7 = K ko..ko+7,
// elems 8..15 = K 16+ko..16+ko+7. Source row-major: element (m,k)=base[m*ld+k].
static __device__ __forceinline__ v16h load_a16(const h16* base, int ld) {
  const int lane = threadIdx.x & 31;
  const int m  = lane & 15;
  const int ko = (lane >> 4) << 3;
  const h16* p = base + m * ld + ko;        // 32-bit offset math -> saddr form
  v8h lo = *(const v8h*)(p);
  v8h hi = *(const v8h*)(p + 16);
  v16h r;
#pragma unroll
  for (int i = 0; i < 8; ++i) { r[i] = lo[i]; r[8 + i] = hi[i]; }
  return r;
}

// B 32x16 f16: lane L -> col n=L&15; kb=8*(L>>4); elems 0..7 = K kb..kb+7,
// elems 8..15 = K 16+kb..16+kb+7. Source is B^T stored row-major
// (element (k,n) = base[n*ld + k]) -> two 16B loads per lane.
static __device__ __forceinline__ v16h load_bT16(const h16* base, int ld) {
  const int lane = threadIdx.x & 31;
  const int n  = lane & 15;
  const int kb = (lane >> 4) << 3;
  const h16* p = base + n * ld + kb;
  v8h lo = *(const v8h*)(p);
  v8h hi = *(const v8h*)(p + 16);
  v16h r;
#pragma unroll
  for (int i = 0; i < 8; ++i) { r[i] = lo[i]; r[8 + i] = hi[i]; }
  return r;
}

// ---- One-shot conversion kernels ------------------------------------------
// x f32 -> f16 (vectorized 4-wide, coalesced)
__global__ void xcvt_kernel(const float* __restrict__ x, h16* __restrict__ xH) {
  const int idx = (blockIdx.x * blockDim.x + threadIdx.x) * 4;
  v4f v = *(const v4f*)(x + idx);
  v4h o;
#pragma unroll
  for (int i = 0; i < 4; ++i) o[i] = (h16)v[i];
  *(v4h*)(xH + idx) = o;
}

// W [K,N] f32 -> WT [N,K] f16 (coalesced stores; K,N powers of two)
__global__ void wcvt_kernel(const float* __restrict__ W, h16* __restrict__ WT,
                            int K, int N) {
  const int idx = blockIdx.x * blockDim.x + threadIdx.x;  // over N*K outputs
  const int n = idx / K;
  const int k = idx - n * K;
  WT[idx] = (h16)W[k * N + n];
}

// ---- Kernel 1: projections xH[M,256] * WT[N,256]^T -> f16 ------------------
// transposed==1 stores h as hT[b][channel][token] for contiguous attn B-frags.
__global__ void proj_kernel(const h16* __restrict__ X, const h16* __restrict__ WT,
                            h16* __restrict__ out, int Ncols, int transposed) {
  const int m0 = blockIdx.x * 16;
  const int n0 = blockIdx.y * 16;
  v8f acc = {};
#pragma unroll
  for (int k = 0; k < CCH; k += 32) {
    v16h a = load_a16(X + (size_t)m0 * CCH + k, CCH);
    v16h b = load_bT16(WT + n0 * CCH + k, CCH);   // WT row n = column n of W
    acc = wmma_f16(a, b, acc);
  }
  const int lane = threadIdx.x & 31;
  const int n = lane & 15;
  const int hs = lane >> 4;
#pragma unroll
  for (int v = 0; v < 8; ++v) {
    int m = m0 + v + 8 * hs;      // global token row
    int c = n0 + n;               // output column
    if (transposed) {
      int bb  = m >> 12;          // / NTOK
      int tok = m & (NTOK - 1);
      out[((size_t)bb * CCH + c) * NTOK + tok] = (h16)acc[v];
    } else {
      out[(size_t)m * Ncols + c] = (h16)acc[v];
    }
  }
}

// ---- Kernel 2: flash attention, one wave per 16-query tile -----------------
// 64-key chunks: fixed per-chunk softmax costs (cross-lane reductions,
// 128-reg accumulator rescale, LDS round trip) amortized over 2x the keys.
__global__ void attn_kernel(const h16* __restrict__ fH, const h16* __restrict__ gH,
                            const h16* __restrict__ hT, h16* __restrict__ oH) {
  const int tile = blockIdx.x;            // 0 .. MTOT/16-1
  const int b    = tile >> 8;             // / (NTOK/16)
  const int q0   = (tile & 255) << 4;
  const h16* fB = fH + (size_t)b * NTOK * CDM;
  const h16* gB = gH + (size_t)b * NTOK * CDM;
  const h16* hB = hT + (size_t)b * CCH * NTOK;

  __shared__ __align__(16) h16 pLDS[16 * 64];

  const int lane = threadIdx.x & 31;
  const int n  = lane & 15;
  const int hs = lane >> 4;

  // queries: A fragment 16x32 (d = 32, exactly one K-step)
  const v16h ga = load_a16(gB + q0 * CDM, CDM);

  v8f acc[16];
  const v8f vz = {};
#pragma unroll
  for (int t = 0; t < 16; ++t) acc[t] = vz;
  float mrow[8], lrow[8];
#pragma unroll
  for (int v = 0; v < 8; ++v) { mrow[v] = -1e30f; lrow[v] = 0.0f; }

  for (int k0 = 0; k0 < NTOK; k0 += 64) {
    // Prefetch next chunk: f (4KB contiguous) and the next 128B line of each
    // of the 256 h rows into the WGP caches while this chunk's WMMAs run.
    if (k0 + 64 < NTOK) {
      __builtin_prefetch(fB + (k0 + 64) * CDM + lane * 64, 0, 3);
#pragma unroll
      for (int j = 0; j < 8; ++j)
        __builtin_prefetch(hB + (lane + 32 * j) * NTOK + k0 + 64, 0, 3);
    }

    // s = g . f^T  (four 16x16 key tiles)
    v16h fb0 = load_bT16(fB + (k0     ) * CDM, CDM);
    v16h fb1 = load_bT16(fB + (k0 + 16) * CDM, CDM);
    v16h fb2 = load_bT16(fB + (k0 + 32) * CDM, CDM);
    v16h fb3 = load_bT16(fB + (k0 + 48) * CDM, CDM);
    v8f s0 = wmma_f16(ga, fb0, vz);
    v8f s1 = wmma_f16(ga, fb1, vz);
    v8f s2 = wmma_f16(ga, fb2, vz);
    v8f s3 = wmma_f16(ga, fb3, vz);

    // online softmax: C-layout row m = v + 8*hs lives in VGPR v across the
    // 16 lanes of each half -> reduce with shfl_xor 1,2,4,8.
    v8f sv;                                  // per-row rescale factors
#pragma unroll
    for (int v = 0; v < 8; ++v) {
      float t = fmaxf(fmaxf(s0[v], s1[v]), fmaxf(s2[v], s3[v]));
      t = fmaxf(t, __shfl_xor(t, 1, 32));
      t = fmaxf(t, __shfl_xor(t, 2, 32));
      t = fmaxf(t, __shfl_xor(t, 4, 32));
      t = fmaxf(t, __shfl_xor(t, 8, 32));
      float mnew  = fmaxf(mrow[v], t);
      float scale = __expf(mrow[v] - mnew);
      float e0 = __expf(s0[v] - mnew);
      float e1 = __expf(s1[v] - mnew);
      float e2 = __expf(s2[v] - mnew);
      float e3 = __expf(s3[v] - mnew);
      float ps = (e0 + e1) + (e2 + e3);
      ps += __shfl_xor(ps, 1, 32);
      ps += __shfl_xor(ps, 2, 32);
      ps += __shfl_xor(ps, 4, 32);
      ps += __shfl_xor(ps, 8, 32);
      lrow[v] = lrow[v] * scale + ps;
      mrow[v] = mnew;
      sv[v]   = scale;
      const int m = v + 8 * hs;
      pLDS[m * 64 + n]      = (h16)e0;   // keys k0    .. k0+15
      pLDS[m * 64 + 16 + n] = (h16)e1;   // keys k0+16 .. k0+31
      pLDS[m * 64 + 32 + n] = (h16)e2;   // keys k0+32 .. k0+47
      pLDS[m * 64 + 48 + n] = (h16)e3;   // keys k0+48 .. k0+63
    }
    // rescale accumulators as whole vectors (packs into v_pk_mul_f32)
#pragma unroll
    for (int t2 = 0; t2 < 16; ++t2) acc[t2] *= sv;

    // p (16 queries x 64 keys) re-shaped through LDS into two A fragments;
    // DS ops from one wave are in-order, compiler inserts s_wait_dscnt.
    v16h pa0 = load_a16(pLDS, 64);
    v16h pa1 = load_a16(pLDS + 32, 64);

    // o += p . h_chunk over all 16 channel tiles (hT gives contiguous B-frags;
    // each h row contributes one full 128B cacheline per chunk).
#pragma unroll
    for (int ct = 0; ct < 16; ++ct) {
      v16h hb0 = load_bT16(hB + (ct * 16) * NTOK + k0, NTOK);
      acc[ct] = wmma_f16(pa0, hb0, acc[ct]);
      v16h hb1 = load_bT16(hB + (ct * 16) * NTOK + k0 + 32, NTOK);
      acc[ct] = wmma_f16(pa1, hb1, acc[ct]);
    }
  }

  // normalize and store o (f16)
  v8f inv;
#pragma unroll
  for (int v = 0; v < 8; ++v) inv[v] = 1.0f / lrow[v];
#pragma unroll
  for (int ct = 0; ct < 16; ++ct) {
    v8f r = acc[ct] * inv;
#pragma unroll
    for (int v = 0; v < 8; ++v) {
      const int m = q0 + v + 8 * hs;
      oH[((size_t)b * NTOK + m) * CCH + ct * 16 + n] = (h16)r[v];
    }
  }
}

// ---- Kernel 3: y = gamma * (o . Wv) + x ------------------------------------
__global__ void outproj_kernel(const h16* __restrict__ oH, const h16* __restrict__ WvT,
                               const float* __restrict__ x, const float* __restrict__ gamma,
                               float* __restrict__ y) {
  const int m0 = blockIdx.x * 16;
  const int n0 = blockIdx.y * 16;
  v8f acc = {};
#pragma unroll
  for (int k = 0; k < CCH; k += 32) {
    v16h a = load_a16(oH + (size_t)m0 * CCH + k, CCH);
    v16h b = load_bT16(WvT + n0 * CCH + k, CCH);
    acc = wmma_f16(a, b, acc);
  }
  const float gm = gamma[0];
  const int lane = threadIdx.x & 31;
  const int n  = lane & 15;
  const int hs = lane >> 4;
#pragma unroll
  for (int v = 0; v < 8; ++v) {
    const size_t idx = (size_t)(m0 + v + 8 * hs) * CCH + (n0 + n);
    y[idx] = gm * acc[v] + x[idx];
  }
}

extern "C" void kernel_launch(void* const* d_in, const int* in_sizes, int n_in,
                              void* d_out, int out_size, void* d_ws, size_t ws_size,
                              hipStream_t stream) {
  const float* x     = (const float*)d_in[0];
  const float* Wf    = (const float*)d_in[1];
  const float* Wg    = (const float*)d_in[2];
  const float* Wh    = (const float*)d_in[3];
  const float* Wv    = (const float*)d_in[4];
  const float* gamma = (const float*)d_in[5];
  float* y = (float*)d_out;

  // Workspace layout (all f16):
  // f [MTOT,32] | g [MTOT,32] | hT [B,256,N] | o [MTOT,256] | xH [MTOT,256]
  // | WfT [32,256] | WgT [32,256] | WhT [256,256] | WvT [256,256]
  h16* fH  = (h16*)d_ws;
  h16* gH  = fH  + (size_t)MTOT * CDM;
  h16* hT  = gH  + (size_t)MTOT * CDM;
  h16* oH  = hT  + (size_t)MTOT * CCH;
  h16* xH  = oH  + (size_t)MTOT * CCH;
  h16* WfT = xH  + (size_t)MTOT * CCH;
  h16* WgT = WfT + (size_t)CDM * CCH;
  h16* WhT = WgT + (size_t)CDM * CCH;
  h16* WvT = WhT + (size_t)CCH * CCH;

  dim3 blk(32);      // one wave32 per block for WMMA kernels
  dim3 cblk(256);

  // one-shot conversions
  xcvt_kernel<<<dim3((MTOT * CCH / 4) / 256), cblk, 0, stream>>>(x, xH);
  wcvt_kernel<<<dim3((CCH * CDM) / 256), cblk, 0, stream>>>(Wf, WfT, CCH, CDM);
  wcvt_kernel<<<dim3((CCH * CDM) / 256), cblk, 0, stream>>>(Wg, WgT, CCH, CDM);
  wcvt_kernel<<<dim3((CCH * CCH) / 256), cblk, 0, stream>>>(Wh, WhT, CCH, CCH);
  wcvt_kernel<<<dim3((CCH * CCH) / 256), cblk, 0, stream>>>(Wv, WvT, CCH, CCH);

  // projections
  proj_kernel<<<dim3(MTOT / 16, CDM / 16), blk, 0, stream>>>(xH, WfT, fH, CDM, 0);
  proj_kernel<<<dim3(MTOT / 16, CDM / 16), blk, 0, stream>>>(xH, WgT, gH, CDM, 0);
  proj_kernel<<<dim3(MTOT / 16, CCH / 16), blk, 0, stream>>>(xH, WhT, hT, CCH, 1);

  // flash attention
  attn_kernel<<<dim3(MTOT / 16), blk, 0, stream>>>(fH, gH, hT, oH);

  // output projection + residual
  outproj_kernel<<<dim3(MTOT / 16, CCH / 16), blk, 0, stream>>>(oH, WvT, x, gamma, y);
}
